// CausalSelfAttention_56753697849991
// MI455X (gfx1250) — compile-verified, matrix operand order checked by reading, not run on hardware
//
#include <hip/hip_runtime.h>

typedef __bf16 bf16_t;
typedef __attribute__((ext_vector_type(16))) __bf16 v16bf;
typedef __attribute__((ext_vector_type(8)))  __bf16 v8bf;
typedef __attribute__((ext_vector_type(8)))  float  v8f;

#define DEVINL __device__ __forceinline__

static constexpr int Bb  = 2;
static constexpr int S   = 2048;
static constexpr int E   = 1024;
static constexpr int H   = 16;
static constexpr int Dh  = 64;
static constexpr int Mrows = Bb * S;      // 4096
static constexpr int NQKV  = 3 * E;       // 3072

// ---------------------------------------------------------------------------
// 16-bit WMMA fragment loader (A or B^T), per CDNA5 ISA 7.12.2:
// lane l holds row (l&15); element e: k = 8*(l>>4) + e      (e = 0..7)
//                                     k = 16 + 8*(l>>4)+e-8 (e = 8..15)
// -> two contiguous 16-byte loads per lane.
// ---------------------------------------------------------------------------
DEVINL v16bf load_frag(const bf16_t* __restrict__ base, int stride, int row,
                       int k0, int lane) {
  const int hf = lane >> 4;
  const bf16_t* p = base + (size_t)row * stride + k0 + 8 * hf;
  v8bf lo = *(const v8bf*)(p);
  v8bf hi = *(const v8bf*)(p + 16);
  return __builtin_shufflevector(lo, hi, 0, 1, 2, 3, 4, 5, 6, 7,
                                         8, 9, 10, 11, 12, 13, 14, 15);
}

DEVINL v8f wmma_bf16(v16bf a, v16bf b, v8f c) {
  return __builtin_amdgcn_wmma_f32_16x16x32_bf16(
      /*neg_a=*/false, a, /*neg_b=*/false, b,
      /*c_mod=*/(short)0, c, /*reuse_a=*/false, /*reuse_b=*/false);
}

// ---------------------------------------------------------------------------
// fp32 -> bf16 elementwise convert
// ---------------------------------------------------------------------------
__global__ void __launch_bounds__(256)
cvt_bf16_kernel(const float* __restrict__ src, bf16_t* __restrict__ dst, int n) {
  int i = blockIdx.x * 256 + threadIdx.x;
  if (i < n) dst[i] = (bf16_t)src[i];
}

// src [K,N] fp32 -> dst [N,K] bf16 (transpose + convert)
__global__ void __launch_bounds__(256)
cvt_transpose_kernel(const float* __restrict__ src, bf16_t* __restrict__ dst,
                     int K, int N) {
  int i = blockIdx.x * 256 + threadIdx.x;
  if (i < K * N) {
    int k = i / N, n = i - k * N;
    dst[(size_t)n * K + k] = (bf16_t)src[i];
  }
}

// ---------------------------------------------------------------------------
// QKV projection: [4096,1024] x [1024,3072] + bias.
// Each wave owns a 32x64 output tile; fragments software-pipelined (preload
// k+32 into distinct registers while k's 8 WMMAs run; last step peeled).
// Epilogue scatters into Q,K:[B,H,S,D] and V^T:[B,H,D,S] (all bf16).
// grid: (NQKV/512, Mrows/32), block 256 (8 waves, wave w owns 64 cols).
// ---------------------------------------------------------------------------
__global__ void __launch_bounds__(256)
qkv_gemm_kernel(const bf16_t* __restrict__ xb, const bf16_t* __restrict__ wT,
                const float* __restrict__ bias, bf16_t* __restrict__ Qb,
                bf16_t* __restrict__ Kb, bf16_t* __restrict__ Vt) {
  const int lane = threadIdx.x & 31;
  const int w    = threadIdx.x >> 5;
  const int ln   = lane & 15;
  const int hf   = lane >> 4;
  const int m0   = blockIdx.y * 32;
  const int n0   = blockIdx.x * 512 + w * 64;

  v8f acc[2][4];
#pragma unroll
  for (int mi = 0; mi < 2; ++mi)
#pragma unroll
    for (int t = 0; t < 4; ++t) acc[mi][t] = v8f{};

  v16bf a0 = load_frag(xb, E, m0 + ln,      0, lane);
  v16bf a1 = load_frag(xb, E, m0 + 16 + ln, 0, lane);
  v16bf b0 = load_frag(wT, E, n0 +  0 + ln, 0, lane);
  v16bf b1 = load_frag(wT, E, n0 + 16 + ln, 0, lane);
  v16bf b2 = load_frag(wT, E, n0 + 32 + ln, 0, lane);
  v16bf b3 = load_frag(wT, E, n0 + 48 + ln, 0, lane);

  for (int k0 = 0; k0 < E - 32; k0 += 32) {
    const int kn = k0 + 32;
    v16bf a0n = load_frag(xb, E, m0 + ln,      kn, lane);
    v16bf a1n = load_frag(xb, E, m0 + 16 + ln, kn, lane);
    v16bf b0n = load_frag(wT, E, n0 +  0 + ln, kn, lane);
    v16bf b1n = load_frag(wT, E, n0 + 16 + ln, kn, lane);
    v16bf b2n = load_frag(wT, E, n0 + 32 + ln, kn, lane);
    v16bf b3n = load_frag(wT, E, n0 + 48 + ln, kn, lane);

    acc[0][0] = wmma_bf16(a0, b0, acc[0][0]);
    acc[1][0] = wmma_bf16(a1, b0, acc[1][0]);
    acc[0][1] = wmma_bf16(a0, b1, acc[0][1]);
    acc[1][1] = wmma_bf16(a1, b1, acc[1][1]);
    acc[0][2] = wmma_bf16(a0, b2, acc[0][2]);
    acc[1][2] = wmma_bf16(a1, b2, acc[1][2]);
    acc[0][3] = wmma_bf16(a0, b3, acc[0][3]);
    acc[1][3] = wmma_bf16(a1, b3, acc[1][3]);

    a0 = a0n; a1 = a1n; b0 = b0n; b1 = b1n; b2 = b2n; b3 = b3n;
  }
  acc[0][0] = wmma_bf16(a0, b0, acc[0][0]);
  acc[1][0] = wmma_bf16(a1, b0, acc[1][0]);
  acc[0][1] = wmma_bf16(a0, b1, acc[0][1]);
  acc[1][1] = wmma_bf16(a1, b1, acc[1][1]);
  acc[0][2] = wmma_bf16(a0, b2, acc[0][2]);
  acc[1][2] = wmma_bf16(a1, b2, acc[1][2]);
  acc[0][3] = wmma_bf16(a0, b3, acc[0][3]);
  acc[1][3] = wmma_bf16(a1, b3, acc[1][3]);

#pragma unroll
  for (int mi = 0; mi < 2; ++mi) {
#pragma unroll
    for (int t = 0; t < 4; ++t) {
      union { v8f v; float f[8]; } u; u.v = acc[mi][t];
      const int n = n0 + t * 16 + ln;
      const float bv = bias[n];
#pragma unroll
      for (int j = 0; j < 8; ++j) {
        int m = m0 + mi * 16 + 8 * hf + j;
        int b = m >> 11;           // / S
        int s = m & (S - 1);
        bf16_t val = (bf16_t)(u.f[j] + bv);
        if (n < E) {
          int hh = n >> 6, d = n & 63;
          Qb[((size_t)(b * H + hh) * S + s) * Dh + d] = val;
        } else if (n < 2 * E) {
          int n2 = n - E; int hh = n2 >> 6, d = n2 & 63;
          Kb[((size_t)(b * H + hh) * S + s) * Dh + d] = val;
        } else {
          int n2 = n - 2 * E; int hh = n2 >> 6, d = n2 & 63;
          Vt[((size_t)(b * H + hh) * Dh + d) * S + s] = val;
        }
      }
    }
  }
}

// ---------------------------------------------------------------------------
// Flash attention: one wave per (b, h, 32-query tile). Streams 32-key tiles.
// Each K-fragment feeds 2 score WMMAs and each V-fragment 2 PV WMMAs (shared
// across the two 16-row subtiles). Online softmax via shfl_xor over 16-lane
// halves; P relaid out through a wave-private LDS tile; next key tile
// prefetched with global_prefetch. Output y -> [B*S, E] bf16.
// ---------------------------------------------------------------------------
__global__ void __launch_bounds__(256)
attn_kernel(const bf16_t* __restrict__ Qb, const bf16_t* __restrict__ Kb,
            const bf16_t* __restrict__ Vt, bf16_t* __restrict__ Yb) {
  __shared__ __align__(16) bf16_t plds[8][32 * 32];

  const int lane = threadIdx.x & 31;
  const int w    = threadIdx.x >> 5;
  const int ln   = lane & 15;
  const int hf   = lane >> 4;

  const int gw = blockIdx.x * 8 + w;       // 0..2047
  const int qt = gw & (S / 32 - 1);        // query tile (32 rows), 0..63
  const int bh = gw >> 6;                  // 0..31 (b*H + h)

  const bf16_t* Qh = Qb + (size_t)bh * S * Dh;
  const bf16_t* Kh = Kb + (size_t)bh * S * Dh;
  const bf16_t* Vh = Vt + (size_t)bh * Dh * S;

  v16bf qa[2][2];
#pragma unroll
  for (int mi = 0; mi < 2; ++mi) {
    qa[mi][0] = load_frag(Qh, Dh, qt * 32 + mi * 16 + ln, 0,  lane);
    qa[mi][1] = load_frag(Qh, Dh, qt * 32 + mi * 16 + ln, 32, lane);
  }

  union { v8f v; float f[8]; } o[2][4];
  float rowmax[2][8], rowsum[2][8];
#pragma unroll
  for (int mi = 0; mi < 2; ++mi) {
#pragma unroll
    for (int t = 0; t < 4; ++t) o[mi][t].v = v8f{};
#pragma unroll
    for (int j = 0; j < 8; ++j) { rowmax[mi][j] = -INFINITY; rowsum[mi][j] = 0.f; }
  }

  const int nkt = qt + 1;                  // causal: 32-key tiles

  for (int kt = 0; kt < nkt; ++kt) {
    const int kbase = kt * 32;

    v16bf kf[2][2];
#pragma unroll
    for (int ks = 0; ks < 2; ++ks) {
      kf[ks][0] = load_frag(Kh, Dh, kbase + ks * 16 + ln, 0,  lane);
      kf[ks][1] = load_frag(Kh, Dh, kbase + ks * 16 + ln, 32, lane);
    }
    if (kt + 1 < nkt) {  // prefetch next key tile rows (global_prefetch_b8)
      __builtin_prefetch(Kh + (size_t)(kbase + 32 + ln) * Dh, 0, 1);
      __builtin_prefetch(Kh + (size_t)(kbase + 48 + ln) * Dh, 0, 1);
    }

    union { v8f v; float f[8]; } sc[2][2];
#pragma unroll
    for (int mi = 0; mi < 2; ++mi)
#pragma unroll
      for (int ks = 0; ks < 2; ++ks) {
        v8f sv = v8f{};
        sv = wmma_bf16(qa[mi][0], kf[ks][0], sv);
        sv = wmma_bf16(qa[mi][1], kf[ks][1], sv);
        sc[mi][ks].v = sv;
      }

    float corr[2][8];
#pragma unroll
    for (int mi = 0; mi < 2; ++mi) {
      const int key0 = kbase + ln, key1 = key0 + 16;
      float mx[8];
#pragma unroll
      for (int j = 0; j < 8; ++j) {
        int row = qt * 32 + mi * 16 + 8 * hf + j;
        float a = sc[mi][0].f[j] * 0.125f; if (key0 > row) a = -INFINITY;
        float b = sc[mi][1].f[j] * 0.125f; if (key1 > row) b = -INFINITY;
        sc[mi][0].f[j] = a; sc[mi][1].f[j] = b;
        mx[j] = fmaxf(a, b);
      }
#pragma unroll
      for (int off = 1; off < 16; off <<= 1)
#pragma unroll
        for (int j = 0; j < 8; ++j)
          mx[j] = fmaxf(mx[j], __shfl_xor(mx[j], off, 32));

      float ps[8];
#pragma unroll
      for (int j = 0; j < 8; ++j) {
        float mn = fmaxf(rowmax[mi][j], mx[j]);
        corr[mi][j] = __expf(rowmax[mi][j] - mn);
        rowmax[mi][j] = mn;
        float p0 = __expf(sc[mi][0].f[j] - mn);
        float p1 = __expf(sc[mi][1].f[j] - mn);
        sc[mi][0].f[j] = p0; sc[mi][1].f[j] = p1;
        ps[j] = p0 + p1;
      }
#pragma unroll
      for (int off = 1; off < 16; off <<= 1)
#pragma unroll
        for (int j = 0; j < 8; ++j)
          ps[j] += __shfl_xor(ps[j], off, 32);
#pragma unroll
      for (int j = 0; j < 8; ++j)
        rowsum[mi][j] = rowsum[mi][j] * corr[mi][j] + ps[j];

      // P subtile -> LDS (C-layout out)
      bf16_t* pl = &plds[w][mi * 16 * 32];
#pragma unroll
      for (int j = 0; j < 8; ++j) {
        pl[(8 * hf + j) * 32 + ln]      = (bf16_t)sc[mi][0].f[j];
        pl[(8 * hf + j) * 32 + 16 + ln] = (bf16_t)sc[mi][1].f[j];
      }
    }
    asm volatile("" ::: "memory");   // LDS write->read order (in-order DS, same wave)
    v16bf pf[2];
    pf[0] = load_frag(&plds[w][0],       32, ln, 0, lane);
    pf[1] = load_frag(&plds[w][16 * 32], 32, ln, 0, lane);
    asm volatile("" ::: "memory");

    // rescale accumulators once per key tile
#pragma unroll
    for (int mi = 0; mi < 2; ++mi)
#pragma unroll
      for (int t = 0; t < 4; ++t)
#pragma unroll
        for (int j = 0; j < 8; ++j) o[mi][t].f[j] *= corr[mi][j];

#pragma unroll
    for (int t = 0; t < 4; ++t) {
      v16bf vf = load_frag(Vh, S, t * 16 + ln, kbase, lane);
      o[0][t].v = wmma_bf16(pf[0], vf, o[0][t].v);
      o[1][t].v = wmma_bf16(pf[1], vf, o[1][t].v);
    }
  }

  const int b  = bh >> 4;
  const int hh = bh & 15;
#pragma unroll
  for (int mi = 0; mi < 2; ++mi)
#pragma unroll
    for (int t = 0; t < 4; ++t)
#pragma unroll
      for (int j = 0; j < 8; ++j) {
        int s = qt * 32 + mi * 16 + 8 * hf + j;
        float val = o[mi][t].f[j] / rowsum[mi][j];
        Yb[((size_t)b * S + s) * E + hh * Dh + t * 16 + ln] = (bf16_t)val;
      }
}

// ---------------------------------------------------------------------------
// Output projection: [4096,1024] x [1024,1024] + bias -> fp32 out.
// Same 32x64-per-wave pipelined scheme. grid: (E/512, Mrows/32), block 256.
// ---------------------------------------------------------------------------
__global__ void __launch_bounds__(256)
out_gemm_kernel(const bf16_t* __restrict__ yb, const bf16_t* __restrict__ wT,
                const float* __restrict__ bias, float* __restrict__ out) {
  const int lane = threadIdx.x & 31;
  const int w    = threadIdx.x >> 5;
  const int ln   = lane & 15;
  const int hf   = lane >> 4;
  const int m0   = blockIdx.y * 32;
  const int n0   = blockIdx.x * 512 + w * 64;

  v8f acc[2][4];
#pragma unroll
  for (int mi = 0; mi < 2; ++mi)
#pragma unroll
    for (int t = 0; t < 4; ++t) acc[mi][t] = v8f{};

  v16bf a0 = load_frag(yb, E, m0 + ln,      0, lane);
  v16bf a1 = load_frag(yb, E, m0 + 16 + ln, 0, lane);
  v16bf b0 = load_frag(wT, E, n0 +  0 + ln, 0, lane);
  v16bf b1 = load_frag(wT, E, n0 + 16 + ln, 0, lane);
  v16bf b2 = load_frag(wT, E, n0 + 32 + ln, 0, lane);
  v16bf b3 = load_frag(wT, E, n0 + 48 + ln, 0, lane);

  for (int k0 = 0; k0 < E - 32; k0 += 32) {
    const int kn = k0 + 32;
    v16bf a0n = load_frag(yb, E, m0 + ln,      kn, lane);
    v16bf a1n = load_frag(yb, E, m0 + 16 + ln, kn, lane);
    v16bf b0n = load_frag(wT, E, n0 +  0 + ln, kn, lane);
    v16bf b1n = load_frag(wT, E, n0 + 16 + ln, kn, lane);
    v16bf b2n = load_frag(wT, E, n0 + 32 + ln, kn, lane);
    v16bf b3n = load_frag(wT, E, n0 + 48 + ln, kn, lane);

    acc[0][0] = wmma_bf16(a0, b0, acc[0][0]);
    acc[1][0] = wmma_bf16(a1, b0, acc[1][0]);
    acc[0][1] = wmma_bf16(a0, b1, acc[0][1]);
    acc[1][1] = wmma_bf16(a1, b1, acc[1][1]);
    acc[0][2] = wmma_bf16(a0, b2, acc[0][2]);
    acc[1][2] = wmma_bf16(a1, b2, acc[1][2]);
    acc[0][3] = wmma_bf16(a0, b3, acc[0][3]);
    acc[1][3] = wmma_bf16(a1, b3, acc[1][3]);

    a0 = a0n; a1 = a1n; b0 = b0n; b1 = b1n; b2 = b2n; b3 = b3n;
  }
  acc[0][0] = wmma_bf16(a0, b0, acc[0][0]);
  acc[1][0] = wmma_bf16(a1, b0, acc[1][0]);
  acc[0][1] = wmma_bf16(a0, b1, acc[0][1]);
  acc[1][1] = wmma_bf16(a1, b1, acc[1][1]);
  acc[0][2] = wmma_bf16(a0, b2, acc[0][2]);
  acc[1][2] = wmma_bf16(a1, b2, acc[1][2]);
  acc[0][3] = wmma_bf16(a0, b3, acc[0][3]);
  acc[1][3] = wmma_bf16(a1, b3, acc[1][3]);

#pragma unroll
  for (int mi = 0; mi < 2; ++mi)
#pragma unroll
    for (int t = 0; t < 4; ++t) {
      union { v8f v; float f[8]; } u; u.v = acc[mi][t];
      const int n = n0 + t * 16 + ln;
      const float bv = bias[n];
#pragma unroll
      for (int j = 0; j < 8; ++j) {
        int m = m0 + mi * 16 + 8 * hf + j;
        out[(size_t)m * E + n] = u.f[j] + bv;
      }
    }
}

// ---------------------------------------------------------------------------
extern "C" void kernel_launch(void* const* d_in, const int* in_sizes, int n_in,
                              void* d_out, int out_size, void* d_ws, size_t ws_size,
                              hipStream_t stream) {
  const float* x     = (const float*)d_in[0];
  const float* W_qkv = (const float*)d_in[1];
  const float* b_qkv = (const float*)d_in[2];
  const float* W_out = (const float*)d_in[3];
  const float* b_out = (const float*)d_in[4];
  float* out = (float*)d_out;

  size_t off = 0;
  auto carve = [&](size_t bytes) -> void* {
    void* p = (char*)d_ws + off;
    off += (bytes + 255) & ~(size_t)255;
    return p;
  };
  bf16_t* xb    = (bf16_t*)carve((size_t)Mrows * E * 2);        // 8 MB
  bf16_t* wqkvT = (bf16_t*)carve((size_t)NQKV * E * 2);         // 6 MB
  bf16_t* woutT = (bf16_t*)carve((size_t)E * E * 2);            // 2 MB
  bf16_t* Qb    = (bf16_t*)carve((size_t)Bb * H * S * Dh * 2);  // 8 MB
  bf16_t* Kb    = (bf16_t*)carve((size_t)Bb * H * S * Dh * 2);  // 8 MB
  bf16_t* Vt    = (bf16_t*)carve((size_t)Bb * H * Dh * S * 2);  // 8 MB
  bf16_t* Yb    = (bf16_t*)carve((size_t)Mrows * E * 2);        // 8 MB

  cvt_bf16_kernel<<<(Mrows * E) / 256, 256, 0, stream>>>(x, xb, Mrows * E);
  cvt_transpose_kernel<<<(E * NQKV) / 256, 256, 0, stream>>>(W_qkv, wqkvT, E, NQKV);
  cvt_transpose_kernel<<<(E * E) / 256, 256, 0, stream>>>(W_out, woutT, E, E);

  qkv_gemm_kernel<<<dim3(NQKV / 512, Mrows / 32), 256, 0, stream>>>(
      xb, wqkvT, b_qkv, Qb, Kb, Vt);

  attn_kernel<<<(Bb * H * (S / 32)) / 8, 256, 0, stream>>>(Qb, Kb, Vt, Yb);

  out_gemm_kernel<<<dim3(E / 512, Mrows / 32), 256, 0, stream>>>(
      Yb, woutT, b_out, out);
}
